// MessageLayer_22926535426528
// MI455X (gfx1250) — matrix operand order repaired and checked by reading, not compiled, not run on hardware
//
#include <hip/hip_runtime.h>
#include <cstdint>
#include <cstddef>

typedef _Float16 half_t;
typedef __attribute__((ext_vector_type(4)))  _Float16 v4h;
typedef __attribute__((ext_vector_type(8)))  _Float16 v8h;
typedef __attribute__((ext_vector_type(16))) _Float16 v16h;
typedef __attribute__((ext_vector_type(8)))  float    v8f;

// ---------------------------------------------------------------------------
// helpers
// ---------------------------------------------------------------------------
__device__ __forceinline__ v16h combine16(v8h lo, v8h hi) {
  v16h r;
#pragma unroll
  for (int i = 0; i < 8; ++i) { r[i] = lo[i]; r[i + 8] = hi[i]; }
  return r;
}

// A fragment for V_WMMA_F32_16X16X32_F16 from a row-major f16 tile [16][128].
// ISA layout: lanes 0-15 (M=lane) hold K = c*32 + {0..7, 16..23},
//             lanes 16-31 (M=lane-16) hold K = c*32 + {8..15, 24..31}.
__device__ __forceinline__ v16h load_a_frag(const half_t* tile, int nl, int g, int c) {
  const half_t* p = tile + nl * 128 + c * 32 + g;          // g = 0 or 8
  return combine16(*(const v8h*)p, *(const v8h*)(p + 16)); // two runs of 8 halves
}

// B fragment from a fragment-major packed weight buffer (16 contiguous halves
// per lane, fully coalesced 32B/lane global load).
__device__ __forceinline__ v16h load_b_frag(const half_t* pack, int ntiles,
                                            int c, int j, int lane) {
  const half_t* p = pack + (size_t)((c * ntiles + j) * 32 + lane) * 16;
  return combine16(*(const v8h*)p, *(const v8h*)(p + 8));
}

// float atomic max via sign-aware integer ordering (native int atomics).
__device__ __forceinline__ void atomic_max_f32(float* addr, float val) {
  if (val >= 0.0f) atomicMax((int*)addr, __float_as_int(val));
  else             atomicMin((unsigned int*)addr, __float_as_uint(val));
}

// ---------------------------------------------------------------------------
// kernel 0: init workspace (every launch -> graph-replay safe)
// ---------------------------------------------------------------------------
__global__ void init_ws_kernel(float* __restrict__ gmax, float* __restrict__ gsum,
                               float* __restrict__ accum, int N) {
  int i = blockIdx.x * blockDim.x + threadIdx.x;
  if (i < N * 64) accum[i] = 0.0f;
  if (i < N) { gmax[i] = __int_as_float(0xFF800000); gsum[i] = 0.0f; }
}

// ---------------------------------------------------------------------------
// kernel 0b: pack f32 weights into fragment-major f16 buffers in ws
//   pg1/pm1: [c=4][j=8][lane=32][h=16]  from [128][128]
//   pm2    : [c=4][j=4][lane=32][h=16]  from [128][64]
//   B layout: n = j*16 + (lane&15), k = c*32 + (lane<16 ? 0 : 8) + (h&7) + (h&8 ? 16 : 0)
// ---------------------------------------------------------------------------
__global__ void pack_weights_kernel(const float* __restrict__ gw1,
                                    const float* __restrict__ mw1,
                                    const float* __restrict__ mw2,
                                    half_t* __restrict__ pg1,
                                    half_t* __restrict__ pm1,
                                    half_t* __restrict__ pm2) {
  int idx = blockIdx.x * blockDim.x + threadIdx.x;
  if (idx < 16384) {
    int h = idx & 15;
    int l = (idx >> 4) & 31;
    int t = idx >> 9;               // c*8 + j
    int j = t & 7, c = t >> 3;
    int n = j * 16 + (l & 15);
    int k = c * 32 + ((l < 16) ? 0 : 8) + (h & 7) + ((h & 8) ? 16 : 0);
    pg1[idx] = (half_t)gw1[k * 128 + n];
    pm1[idx] = (half_t)mw1[k * 128 + n];
  }
  if (idx < 8192) {
    int h = idx & 15;
    int l = (idx >> 4) & 31;
    int t = idx >> 9;               // c*4 + j
    int j = t & 3, c = t >> 2;
    int n = j * 16 + (l & 15);
    int k = c * 32 + ((l < 16) ? 0 : 8) + (h & 7) + ((h & 8) ? 16 : 0);
    pm2[idx] = (half_t)mw2[k * 64 + n];
  }
}

// ---------------------------------------------------------------------------
// kernel 1: gate MLP per 16-edge tile (one wave per tile), WMMA first layer,
// shuffle-reduced dot for the 128->1 second layer; logits -> ws, atomicMax.
// ---------------------------------------------------------------------------
__global__ __launch_bounds__(256) void gate_pass_kernel(
    const float* __restrict__ feats,
    const int* __restrict__ self_idx, const int* __restrict__ nbr_idx,
    const half_t* __restrict__ pg1,
    const float* __restrict__ gb1, const float* __restrict__ gw2,
    const float* __restrict__ gb2,
    float* __restrict__ logits, float* __restrict__ gmax, int E) {
  __shared__ half_t msgT[8 * 2048];   // 8 waves x [16 edges][128] f16
  __shared__ float gb1s[128];
  __shared__ float gw2s[128];
  for (int i = threadIdx.x; i < 128; i += blockDim.x) { gb1s[i] = gb1[i]; gw2s[i] = gw2[i]; }
  __syncthreads();

  const int wid  = threadIdx.x >> 5;
  const int lane = threadIdx.x & 31;
  const int nl   = lane & 15;
  const bool hi  = lane >= 16;
  const int e0   = (blockIdx.x * 8 + wid) * 16;
  if (e0 >= E) return;
  half_t* tile = msgT + wid * 2048;

  // gather + f16-convert msg tile: cols 0..63 = self feats, 64..127 = neighbor
#pragma unroll
  for (int r = 0; r < 16; ++r) {
    int e = e0 + r; if (e >= E) e = E - 1;
    int src = hi ? nbr_idx[e] : self_idx[e];
    const float4 v = *(const float4*)(feats + (size_t)src * 64 + nl * 4);
    v4h hv = { (half_t)v.x, (half_t)v.y, (half_t)v.z, (half_t)v.w };
    *(v4h*)(tile + r * 128 + (hi ? 64 : 0) + nl * 4) = hv;
  }

  const int g = hi ? 8 : 0;
  v16h afr[4];
#pragma unroll
  for (int c = 0; c < 4; ++c) afr[c] = load_a_frag(tile, nl, g, c);

  float gval[8];
#pragma unroll
  for (int r = 0; r < 8; ++r) gval[r] = 0.0f;

#pragma unroll
  for (int j = 0; j < 8; ++j) {              // 8 N-tiles of H=128
    v8f acc = {};
#pragma unroll
    for (int c = 0; c < 4; ++c) {            // K = 128 in 4 chunks of 32
      v16h b = load_b_frag(pg1, 8, c, j, lane);
      acc = __builtin_amdgcn_wmma_f32_16x16x32_f16(false, afr[c], false, b,
                                                   (short)0, acc, false, false);
    }
    float bias = gb1s[j * 16 + nl];
    float wv   = gw2s[j * 16 + nl];
#pragma unroll
    for (int r = 0; r < 8; ++r) {            // relu + fold 128->1 layer
      float hval = acc[r] + bias;
      gval[r] += (hval > 0.0f ? hval : 0.0f) * wv;
    }
  }

  const float gb2v = gb2[0];
#pragma unroll
  for (int r = 0; r < 8; ++r) {              // reduce across 16-lane halves
    float v = gval[r];
    v += __shfl_xor(v, 1, 32);
    v += __shfl_xor(v, 2, 32);
    v += __shfl_xor(v, 4, 32);
    v += __shfl_xor(v, 8, 32);
    if (nl == r) {                           // lanes r and 16+r own edges r, r+8
      int e = e0 + r + (hi ? 8 : 0);
      if (e < E) {
        float logit = v + gb2v;
        logits[e] = logit;
        atomic_max_f32(&gmax[self_idx[e]], logit);
      }
    }
  }
}

// ---------------------------------------------------------------------------
// kernel 2: message MLP (both layers WMMA) + weighted-exp gate + scatter-add
// ---------------------------------------------------------------------------
__global__ __launch_bounds__(256) void msg_pass_kernel(
    const float* __restrict__ node_w, const float* __restrict__ feats,
    const int* __restrict__ self_idx, const int* __restrict__ nbr_idx,
    const half_t* __restrict__ pm1, const half_t* __restrict__ pm2,
    const float* __restrict__ mb1, const float* __restrict__ mb2,
    const float* __restrict__ pow_p,
    const float* __restrict__ logits, const float* __restrict__ gmax,
    float* __restrict__ gsum, float* __restrict__ accum, int E) {
  __shared__ half_t msgT[8 * 2048];
  __shared__ float mb1s[128];
  __shared__ float mb2s[64];
  for (int i = threadIdx.x; i < 128; i += blockDim.x) mb1s[i] = mb1[i];
  for (int i = threadIdx.x; i < 64;  i += blockDim.x) mb2s[i] = mb2[i];
  __syncthreads();

  const int wid  = threadIdx.x >> 5;
  const int lane = threadIdx.x & 31;
  const int nl   = lane & 15;
  const bool hi  = lane >= 16;
  const int e0   = (blockIdx.x * 8 + wid) * 16;
  if (e0 >= E) return;
  half_t* tile = msgT + wid * 2048;

#pragma unroll
  for (int r = 0; r < 16; ++r) {
    int e = e0 + r; if (e >= E) e = E - 1;
    int src = hi ? nbr_idx[e] : self_idx[e];
    const float4 v = *(const float4*)(feats + (size_t)src * 64 + nl * 4);
    v4h hv = { (half_t)v.x, (half_t)v.y, (half_t)v.z, (half_t)v.w };
    *(v4h*)(tile + r * 128 + (hi ? 64 : 0) + nl * 4) = hv;
  }

  const int g = hi ? 8 : 0;
  v16h afr[4];
#pragma unroll
  for (int c = 0; c < 4; ++c) afr[c] = load_a_frag(tile, nl, g, c);

  // layer 1: hidden = relu(msg @ mw1 + mb1); write back to LDS tile as f16 A2
#pragma unroll
  for (int j = 0; j < 8; ++j) {
    v8f acc = {};
#pragma unroll
    for (int c = 0; c < 4; ++c) {
      v16h b = load_b_frag(pm1, 8, c, j, lane);
      acc = __builtin_amdgcn_wmma_f32_16x16x32_f16(false, afr[c], false, b,
                                                   (short)0, acc, false, false);
    }
    float bias = mb1s[j * 16 + nl];
#pragma unroll
    for (int r = 0; r < 8; ++r) {
      float hval = acc[r] + bias;
      hval = hval > 0.0f ? hval : 0.0f;
      tile[(r + (hi ? 8 : 0)) * 128 + j * 16 + nl] = (half_t)hval;  // D-frag -> row-major
    }
  }

  // per-edge gate numerator: p = w^pow * exp(logit - gmax[self]); lanes 0..15
  const float pw = pow_p[0];
  float pval = 0.0f;
  if (!hi) {
    int e = e0 + nl;
    if (e < E) {
      int s = self_idx[e];
      float wv = node_w[nbr_idx[e]];
      pval = powf(wv, pw) * expf(logits[e] - gmax[s]);
      atomicAdd(&gsum[s], pval);
    }
  }

  v16h a2[4];
#pragma unroll
  for (int c = 0; c < 4; ++c) a2[c] = load_a_frag(tile, nl, g, c);

  // layer 2: x = hidden @ mw2 + mb2; scatter p*x into accum[self]
#pragma unroll
  for (int j = 0; j < 4; ++j) {              // 4 N-tiles of D=64
    v8f acc = {};
#pragma unroll
    for (int c = 0; c < 4; ++c) {
      v16h b = load_b_frag(pm2, 4, c, j, lane);
      acc = __builtin_amdgcn_wmma_f32_16x16x32_f16(false, a2[c], false, b,
                                                   (short)0, acc, false, false);
    }
    float bias = mb2s[j * 16 + nl];
#pragma unroll
    for (int r = 0; r < 8; ++r) {
      int m = r + (hi ? 8 : 0);
      int e = e0 + m;
      if (e < E) {
        float pm = __shfl(pval, m, 32);      // broadcast edge m's gate
        float val = (acc[r] + bias) * pm;
        atomicAdd(&accum[(size_t)self_idx[e] * 64 + j * 16 + nl], val);
      }
    }
  }
}

// ---------------------------------------------------------------------------
// kernel 3: out = accum / (gsum + 1e-10) + prev   (division factors out of sum)
// ---------------------------------------------------------------------------
__global__ void finalize_kernel(const float* __restrict__ accum,
                                const float* __restrict__ gsum,
                                const float* __restrict__ prev,
                                float* __restrict__ out, int N) {
  int i = blockIdx.x * blockDim.x + threadIdx.x;
  if (i < N * 64) {
    int n = i >> 6;
    out[i] = accum[i] / (gsum[n] + 1e-10f) + prev[i];
  }
}

// ---------------------------------------------------------------------------
extern "C" void kernel_launch(void* const* d_in, const int* in_sizes, int n_in,
                              void* d_out, int out_size, void* d_ws, size_t ws_size,
                              hipStream_t stream) {
  (void)n_in; (void)out_size; (void)ws_size;
  const float* node_w = (const float*)d_in[0];
  const float* feats  = (const float*)d_in[1];
  const int*   sidx   = (const int*)d_in[2];
  const int*   nidx   = (const int*)d_in[3];
  const float* gw1    = (const float*)d_in[4];
  const float* gb1    = (const float*)d_in[5];
  const float* gw2    = (const float*)d_in[6];
  const float* gb2    = (const float*)d_in[7];
  const float* mw1    = (const float*)d_in[8];
  const float* mb1    = (const float*)d_in[9];
  const float* mw2    = (const float*)d_in[10];
  const float* mb2    = (const float*)d_in[11];
  const float* pow_p  = (const float*)d_in[12];

  const int N = in_sizes[0];   // node_weights flat count == N
  const int E = in_sizes[2];

  // workspace layout
  float* logits = (float*)d_ws;
  float* gmax   = logits + E;
  float* gsum   = gmax + N;
  float* accum  = gsum + N;
  uintptr_t pb  = (uintptr_t)(accum + (size_t)N * 64);
  pb = (pb + 31) & ~(uintptr_t)31;         // 32B align packed weights
  half_t* pg1 = (half_t*)pb;               // 16384 halves
  half_t* pm1 = pg1 + 16384;               // 16384 halves
  half_t* pm2 = pm1 + 16384;               // 8192  halves

  float* out = (float*)d_out;

  const int tot = N * 64;
  init_ws_kernel<<<(tot + 255) / 256, 256, 0, stream>>>(gmax, gsum, accum, N);
  pack_weights_kernel<<<64, 256, 0, stream>>>(gw1, mw1, mw2, pg1, pm1, pm2);

  const int ntiles  = (E + 15) / 16;
  const int nblocks = (ntiles + 7) / 8;    // 8 waves (tiles) per 256-thread block
  gate_pass_kernel<<<nblocks, 256, 0, stream>>>(feats, sidx, nidx, pg1,
                                                gb1, gw2, gb2, logits, gmax, E);
  msg_pass_kernel<<<nblocks, 256, 0, stream>>>(node_w, feats, sidx, nidx,
                                               pm1, pm2, mb1, mb2, pow_p,
                                               logits, gmax, gsum, accum, E);
  finalize_kernel<<<(tot + 255) / 256, 256, 0, stream>>>(accum, gsum, feats, out, N);
}
